// TargetGNN_28681791603120
// MI455X (gfx1250) — matrix operand order repaired and checked by reading, not compiled, not run on hardware
//
#include <hip/hip_runtime.h>
#include <cstdint>

// ---------------- problem constants (match reference) ----------------
#define NN 8000      // nodes
#define EE 64000     // edges
#define FF 2560      // in channels (== H*C, both layers share shapes)
#define HH 8         // heads
#define CC 320       // channels per head
#define HC 2560      // H*C
#define GG 16        // graphs

typedef __bf16 bf16_t;
typedef bf16_t v16bf __attribute__((ext_vector_type(16)));
typedef bf16_t v8bf  __attribute__((ext_vector_type(8)));
typedef float  v8f   __attribute__((ext_vector_type(8)));
typedef int    v4i_vs __attribute__((vector_size(16)));   // matches async builtin param type

#define GLOBAL_AS __attribute__((address_space(1)))
#define LDS_AS    __attribute__((address_space(3)))

#if defined(__gfx1250__) && __has_builtin(__builtin_amdgcn_global_load_async_to_lds_b128)
#define USE_ASYNC_LDS 1
#define ASYNC_CP(gp, lp)                                                     \
  __builtin_amdgcn_global_load_async_to_lds_b128(                            \
      (GLOBAL_AS v4i_vs*)(uintptr_t)(gp), (LDS_AS v4i_vs*)(lp), 0, 0)
#else
#define USE_ASYNC_LDS 0
#endif

static __device__ __forceinline__ void wait_async0() {
#if __has_builtin(__builtin_amdgcn_s_wait_asynccnt)
  __builtin_amdgcn_s_wait_asynccnt(0);
#else
  asm volatile("s_wait_asynccnt 0x0" ::: "memory");
#endif
}

// ---------------- fp32 -> bf16 (RNE) streaming convert (for activations) ----------------
__global__ void k_f32_to_bf16(const float* __restrict__ src,
                              unsigned short* __restrict__ dst, int n) {
  int i = blockIdx.x * blockDim.x + threadIdx.x;
  int stride = gridDim.x * blockDim.x;
  for (; i < n; i += stride) {
    unsigned u = __float_as_uint(src[i]);
    u += 0x7FFFu + ((u >> 16) & 1u);
    dst[i] = (unsigned short)(u >> 16);
  }
}

// ---------------- fp32 W[K][N] -> bf16 WT[N][K] tiled transpose-convert ----------------
__global__ __launch_bounds__(256) void k_w_to_bf16_t(const float* __restrict__ W,
                                                     unsigned short* __restrict__ WT,
                                                     int K, int Nw) {
  __shared__ float tile[32][33];
  int bn = blockIdx.x << 5;                          // N tile origin
  int bk = blockIdx.y << 5;                          // K tile origin
  int tx = threadIdx.x & 31, ty = threadIdx.x >> 5;  // 32 x 8
#pragma unroll
  for (int j = 0; j < 32; j += 8)
    tile[ty + j][tx] = W[(size_t)(bk + ty + j) * Nw + bn + tx];  // coalesced along N
  __syncthreads();
#pragma unroll
  for (int j = 0; j < 32; j += 8) {
    unsigned u = __float_as_uint(tile[tx][ty + j]);              // = W[bk+tx][bn+ty+j]
    u += 0x7FFFu + ((u >> 16) & 1u);
    WT[(size_t)(bn + ty + j) * K + bk + tx] = (unsigned short)(u >> 16);  // coalesced along K
  }
}

// assemble a 16-element bf16 fragment from two aligned 16B LDS chunks
static __device__ __forceinline__ v16bf frag2(const bf16_t* lo, const bf16_t* hi) {
  v8bf a = *reinterpret_cast<const v8bf*>(lo);
  v8bf b = *reinterpret_cast<const v8bf*>(hi);
  return __builtin_shufflevector(a, b, 0,1,2,3,4,5,6,7,8,9,10,11,12,13,14,15);
}

// ---------------- dual-weight WMMA GEMM: O = A(bf16) @ BT(bf16)^T-view + bias ----------------
// A: [M][K] row-major, BT: [Nout][K] row-major (pre-transposed weights)
// block tile 128x128, 8 waves (4M x 2N), wave tile 32x64 -> 2x4 wmma accumulators
__global__ __launch_bounds__(256) void k_gemm_dual(
    const bf16_t* __restrict__ A,
    const bf16_t* __restrict__ BT0, const bf16_t* __restrict__ BT1,
    const float* __restrict__ bias0, const float* __restrict__ bias1,
    float* __restrict__ O0, float* __restrict__ O1,
    int M, int K, int Nout) {
  const bf16_t* BT  = blockIdx.z ? BT1 : BT0;
  const float* bias = blockIdx.z ? bias1 : bias0;
  float* O          = blockIdx.z ? O1 : O0;

  const int LD = 48;  // padded row stride (96B): 16B-aligned fragment chunks, staggered banks
#if USE_ASYNC_LDS
  __shared__ __align__(16) bf16_t As[2][128 * 48];
  __shared__ __align__(16) bf16_t Bs[2][128 * 48];
#else
  __shared__ __align__(16) bf16_t As[1][128 * 48];
  __shared__ __align__(16) bf16_t Bs[1][128 * 48];
#endif

  const int t = threadIdx.x;
  const int lane = t & 31, wv = t >> 5;
  const int wr = wv & 3, wc = wv >> 2;            // wave grid 4(M) x 2(N)
  const int m0 = blockIdx.x * 128, n0 = blockIdx.y * 128;
  const int l15 = lane & 15, half = lane >> 4;

  // loader: each thread owns rows r0 and r0+64 at 16B column chunk c0, for both tiles
  const int r0 = t >> 2;
  const int c0 = (t & 3) << 3;
  int ga0 = m0 + r0;      if (ga0 >= M) ga0 = M - 1;
  int ga1 = m0 + r0 + 64; if (ga1 >= M) ga1 = M - 1;
  const bf16_t* pa0 = A  + (size_t)ga0 * K + c0;
  const bf16_t* pa1 = A  + (size_t)ga1 * K + c0;
  const bf16_t* pb0 = BT + (size_t)(n0 + r0) * K + c0;        // Nout % 128 == 0
  const bf16_t* pb1 = BT + (size_t)(n0 + r0 + 64) * K + c0;
  const int la0 = r0 * LD + c0, la1 = (r0 + 64) * LD + c0;

  v8f acc[2][4];
#pragma unroll
  for (int mi = 0; mi < 2; ++mi)
#pragma unroll
    for (int ni = 0; ni < 4; ++ni) acc[mi][ni] = {};

  const int steps = K >> 5;

#if USE_ASYNC_LDS
  // ---- prologue: async-DMA tile 0 into buffer 0 (ASYNCcnt-tracked)
  ASYNC_CP(pa0, &As[0][la0]);
  ASYNC_CP(pa1, &As[0][la1]);
  ASYNC_CP(pb0, &Bs[0][la0]);
  ASYNC_CP(pb1, &Bs[0][la1]);
#else
  v8bf ra0 = *reinterpret_cast<const v8bf*>(pa0);
  v8bf ra1 = *reinterpret_cast<const v8bf*>(pa1);
  v8bf rb0 = *reinterpret_cast<const v8bf*>(pb0);
  v8bf rb1 = *reinterpret_cast<const v8bf*>(pb1);
#endif

  for (int i = 0; i < steps; ++i) {
#if USE_ASYNC_LDS
    wait_async0();            // this wave's tile-i copies landed
    __syncthreads();          // everyone's tile-i landed; prev compute finished
    if (i + 1 < steps) {      // overlap next tile's DMA with this tile's WMMAs
      const int ko = (i + 1) << 5;
      const int nb = (i + 1) & 1;
      ASYNC_CP(pa0 + ko, &As[nb][la0]);
      ASYNC_CP(pa1 + ko, &As[nb][la1]);
      ASYNC_CP(pb0 + ko, &Bs[nb][la0]);
      ASYNC_CP(pb1 + ko, &Bs[nb][la1]);
    }
    const bf16_t* as = As[i & 1];
    const bf16_t* bs = Bs[i & 1];
#else
    __syncthreads();          // previous compute done, LDS reusable
    *reinterpret_cast<v8bf*>(&As[0][la0]) = ra0;
    *reinterpret_cast<v8bf*>(&As[0][la1]) = ra1;
    *reinterpret_cast<v8bf*>(&Bs[0][la0]) = rb0;
    *reinterpret_cast<v8bf*>(&Bs[0][la1]) = rb1;
    __syncthreads();
    if (i + 1 < steps) {      // issue next tile's loads before compute (latency overlap)
      const int ko = (i + 1) << 5;
      ra0 = *reinterpret_cast<const v8bf*>(pa0 + ko);
      ra1 = *reinterpret_cast<const v8bf*>(pa1 + ko);
      rb0 = *reinterpret_cast<const v8bf*>(pb0 + ko);
      rb1 = *reinterpret_cast<const v8bf*>(pb1 + ko);
    }
    const bf16_t* as = As[0];
    const bf16_t* bs = Bs[0];
#endif

    // fragments: two contiguous 16B LDS chunks per lane (ISA 16-bit layouts)
    v16bf af[2], bfr[4];
#pragma unroll
    for (int mi = 0; mi < 2; ++mi) {
      int base = (wr * 32 + mi * 16 + l15) * LD + (half << 3);   // K: 8*half, 16+8*half
      af[mi] = frag2(&as[base], &as[base + 16]);
    }
#pragma unroll
    for (int ni = 0; ni < 4; ++ni) {
      int base = (wc * 64 + ni * 16 + l15) * LD + (half << 3);   // K: 8*half, 16+8*half
      bfr[ni] = frag2(&bs[base], &bs[base + 16]);
    }
#pragma unroll
    for (int mi = 0; mi < 2; ++mi)
#pragma unroll
      for (int ni = 0; ni < 4; ++ni)
        acc[mi][ni] = __builtin_amdgcn_wmma_f32_16x16x32_bf16(
            false, af[mi], false, bfr[ni], (short)0, acc[mi][ni], false, false);
  }

  // store D (C/D layout: VGPR r -> M = r + 8*half, N = lane&15) + bias
#pragma unroll
  for (int mi = 0; mi < 2; ++mi) {
    int gmBase = m0 + wr * 32 + mi * 16 + (half << 3);
#pragma unroll
    for (int ni = 0; ni < 4; ++ni) {
      int gn = n0 + wc * 64 + ni * 16 + l15;
      float bv = bias[gn];
#pragma unroll
      for (int r = 0; r < 8; ++r) {
        int gm = gmBase + r;
        if (gm < M) O[(size_t)gm * Nout + gn] = acc[mi][ni][r] + bv;
      }
    }
  }
}

// ---------------- attention state init ----------------
__global__ void k_init_attn(unsigned* __restrict__ smax, float* __restrict__ den) {
  int i = blockIdx.x * blockDim.x + threadIdx.x;
  if (i < NN * HH) { smax[i] = 0u; den[i] = 0.f; }   // 0u == -inf in ordered-uint space
}

__global__ void k_init_out(float* __restrict__ out, const float* __restrict__ bias) {
  int i = blockIdx.x * blockDim.x + threadIdx.x;
  int stride = gridDim.x * blockDim.x;
  for (; i < NN * HC; i += stride) out[i] = bias[i % HC];
}

// ---------------- edge scoring: one wave per (edge, head) ----------------
__global__ __launch_bounds__(256) void k_edge_score(
    const float* __restrict__ xl, const float* __restrict__ xr,
    const float* __restrict__ eattr, const int* __restrict__ ei,
    const float* __restrict__ We, const float* __restrict__ att,
    float* __restrict__ score, unsigned* __restrict__ smax) {
  int e = blockIdx.x;
  int h = threadIdx.x >> 5, lane = threadIdx.x & 31;
  int src = ei[e], dst = ei[EE + e];
  float ea = eattr[e];
  const float* pl = xl + (size_t)src * HC + h * CC;
  const float* pr = xr + (size_t)dst * HC + h * CC;
  const float* pw = We + h * CC;
  const float* pa = att + h * CC;
  float s = 0.f;
#pragma unroll
  for (int i = 0; i < CC / 32; ++i) {
    int c = lane + (i << 5);
    float m = pl[c] + pr[c] + ea * pw[c];
    m = (m > 0.f) ? m : 0.2f * m;           // leaky_relu(0.2)
    s += m * pa[c];
  }
  for (int off = 16; off > 0; off >>= 1) s += __shfl_xor(s, off, 32);
  if (lane == 0) {
    score[e * HH + h] = s;
    unsigned u = __float_as_uint(s);
    u = (u & 0x80000000u) ? ~u : (u | 0x80000000u);   // monotone float->uint
    atomicMax(&smax[dst * HH + h], u);
  }
}

// ---------------- exp + denominator ----------------
__global__ void k_edge_exp(const float* __restrict__ score,
                           const unsigned* __restrict__ smax,
                           const int* __restrict__ ei,
                           float* __restrict__ exs, float* __restrict__ den) {
  int idx = blockIdx.x * blockDim.x + threadIdx.x;
  if (idx >= EE * HH) return;
  int e = idx >> 3, h = idx & 7;
  int dst = ei[EE + e];
  unsigned u = smax[dst * HH + h];
  unsigned bits = (u & 0x80000000u) ? (u & 0x7FFFFFFFu) : ~u;
  float ex = __expf(score[idx] - __uint_as_float(bits));
  exs[idx] = ex;
  atomicAdd(&den[dst * HH + h], ex);
}

// ---------------- alpha-weighted aggregation: one wave per (edge, head) ----------------
__global__ __launch_bounds__(256) void k_edge_aggr(
    const float* __restrict__ xl, const float* __restrict__ exs,
    const float* __restrict__ den, const int* __restrict__ ei,
    float* __restrict__ out) {
  int e = blockIdx.x;
  int h = threadIdx.x >> 5, lane = threadIdx.x & 31;
  int src = ei[e], dst = ei[EE + e];
  float alpha = exs[e * HH + h] / den[dst * HH + h];
  const float* pl = xl + (size_t)src * HC + h * CC;
  float* po = out + (size_t)dst * HC + h * CC;
#pragma unroll
  for (int i = 0; i < CC / 32; ++i) {
    int c = lane + (i << 5);
    atomicAdd(&po[c], pl[c] * alpha);
  }
}

// ---------------- global mean pool ----------------
__global__ void k_pool_zero(float* __restrict__ pool, float* __restrict__ cnt) {
  int i = blockIdx.x * blockDim.x + threadIdx.x;
  if (i < GG * HC) pool[i] = 0.f;
  if (i < GG) cnt[i] = 0.f;
}
__global__ void k_pool_cnt(const int* __restrict__ batch, float* __restrict__ cnt) {
  int n = blockIdx.x * blockDim.x + threadIdx.x;
  if (n < NN) atomicAdd(&cnt[batch[n]], 1.f);
}
__global__ void k_pool_sum(const float* __restrict__ hbuf,
                           const int* __restrict__ batch, float* __restrict__ pool) {
  int i = blockIdx.x * blockDim.x + threadIdx.x;
  int stride = gridDim.x * blockDim.x;
  for (; i < NN * HC; i += stride) {
    int n = i / HC;
    atomicAdd(&pool[batch[n] * HC + (i - n * HC)], hbuf[i]);
  }
}
__global__ void k_pool_final(const float* __restrict__ pool,
                             const float* __restrict__ cnt, float* __restrict__ out) {
  int i = blockIdx.x * blockDim.x + threadIdx.x;
  if (i < GG * HC) out[i] = pool[i] / fmaxf(cnt[i / HC], 1.f);
}

// ---------------- host orchestration ----------------
extern "C" void kernel_launch(void* const* d_in, const int* in_sizes, int n_in,
                              void* d_out, int out_size, void* d_ws, size_t ws_size,
                              hipStream_t stream) {
  const float* x     = (const float*)d_in[0];
  const int*   ei    = (const int*)d_in[1];
  const float* eattr = (const float*)d_in[2];
  const int*   batch = (const int*)d_in[3];
  const float* W[2][2] = {{(const float*)d_in[4],  (const float*)d_in[6]},
                          {(const float*)d_in[11], (const float*)d_in[13]}};
  const float* bWl[2] = {(const float*)d_in[5],  (const float*)d_in[12]};
  const float* bWr[2] = {(const float*)d_in[7],  (const float*)d_in[14]};
  const float* We[2]  = {(const float*)d_in[8],  (const float*)d_in[15]};
  const float* att[2] = {(const float*)d_in[9],  (const float*)d_in[16]};
  const float* bia[2] = {(const float*)d_in[10], (const float*)d_in[17]};

  // ---- workspace carve (all 256B aligned)
  char* p = (char*)d_ws;
  auto take = [&](size_t bytes) -> void* {
    void* q = (void*)p;
    p += (bytes + 255) & ~(size_t)255;
    return q;
  };
  float*    XL    = (float*)take(sizeof(float) * (size_t)NN * HC);
  float*    XR    = (float*)take(sizeof(float) * (size_t)NN * HC);
  float*    HBUF  = (float*)take(sizeof(float) * (size_t)NN * HC);
  bf16_t*   XBF   = (bf16_t*)take(2ull * NN * FF);
  bf16_t*   WBL   = (bf16_t*)take(2ull * FF * HC);   // transposed: [N][K]
  bf16_t*   WBR   = (bf16_t*)take(2ull * FF * HC);   // transposed: [N][K]
  float*    SCORE = (float*)take(sizeof(float) * (size_t)EE * HH);
  float*    EXS   = (float*)take(sizeof(float) * (size_t)EE * HH);
  unsigned* SMAX  = (unsigned*)take(sizeof(unsigned) * (size_t)NN * HH);
  float*    DEN   = (float*)take(sizeof(float) * (size_t)NN * HH);
  float*    POOL  = (float*)take(sizeof(float) * (size_t)GG * HC);
  float*    CNT   = (float*)take(sizeof(float) * (size_t)GG);

  const dim3 gemmGrid((NN + 127) / 128, HC / 128, 2);
  const dim3 trGrid(HC / 32, FF / 32);

  auto layer = [&](const float* xin, int l) {
    k_f32_to_bf16<<<2048, 256, 0, stream>>>(xin, (unsigned short*)XBF, NN * FF);
    k_w_to_bf16_t<<<trGrid, 256, 0, stream>>>(W[l][0], (unsigned short*)WBL, FF, HC);
    k_w_to_bf16_t<<<trGrid, 256, 0, stream>>>(W[l][1], (unsigned short*)WBR, FF, HC);
    k_gemm_dual<<<gemmGrid, 256, 0, stream>>>(XBF, WBL, WBR, bWl[l], bWr[l],
                                              XL, XR, NN, FF, HC);
    k_init_attn<<<(NN * HH + 255) / 256, 256, 0, stream>>>(SMAX, DEN);
    k_init_out<<<4096, 256, 0, stream>>>(HBUF, bia[l]);
    k_edge_score<<<EE, 256, 0, stream>>>(XL, XR, eattr, ei, We[l], att[l], SCORE, SMAX);
    k_edge_exp<<<(EE * HH + 255) / 256, 256, 0, stream>>>(SCORE, SMAX, ei, EXS, DEN);
    k_edge_aggr<<<EE, 256, 0, stream>>>(XL, EXS, DEN, ei, HBUF);
  };

  layer(x, 0);        // layer 1: x -> HBUF
  layer(HBUF, 1);     // layer 2: HBUF converted to bf16 first, then HBUF reused as output

  k_pool_zero<<<(GG * HC + 255) / 256, 256, 0, stream>>>(POOL, CNT);
  k_pool_cnt<<<(NN + 255) / 256, 256, 0, stream>>>(batch, CNT);
  k_pool_sum<<<8192, 256, 0, stream>>>(HBUF, batch, POOL);
  k_pool_final<<<(GG * HC + 255) / 256, 256, 0, stream>>>(POOL, CNT, (float*)d_out);
}
